// DSSM_1855425872423
// MI455X (gfx1250) — matrix operand order repaired and checked by reading, not compiled, not run on hardware
//
#include <hip/hip_runtime.h>

// DSS / S4D layer on MI455X (gfx1250): direct causal conv as Toeplitz matmul
// on the WMMA path. H=256 heads, N=64 modes, B=16 batch, L=4096.
//
// Roofline: direct conv = 69 GFLOP in f16 WMMA w/ f32 accum (4.2M
// v_wmma_f32_16x16x32_f16 wave-ops) vs ~134 MB of HBM traffic -> ~6us floor
// at 23.3 TB/s. All conv-phase operands (u16 33.6MB + Kshift 33.8MB) are
// L2-resident (192MB). The 16 lane-shifted copies of the reversed kernel
// make every A fragment two 16B-aligned global_load_b128s, and the lag loop
// trip count is readfirstlane'd into an SGPR so loop control is scalar and
// EXEC is never touched around the WMMA.

#define Hh 256
#define Nn 64
#define Bb 16
#define Ll 4096
#define KS 4128          // shifted-kernel row length (zeros pad both ends)
#define KROW (16 * KS)   // halfs per head in Kshift

typedef __attribute__((ext_vector_type(16))) _Float16 v16h;
typedef __attribute__((ext_vector_type(8)))  _Float16 v8h;
typedef __attribute__((ext_vector_type(4)))  _Float16 v4h;
typedef __attribute__((ext_vector_type(8)))  float    v8f;

// ---------------------------------------------------------------------------
// Phase 1: K[h,l] = 2*Re( sum_n Cs[h,n] * exp(dtA[h,n]*l) ), scattered into
// 16 lane-shifted reversed copies: Kshift[h][m][i] = K[h, 4096 + m - i].
// Unwritten entries (future lags / beyond L) are zero -> causal mask for free.
// ---------------------------------------------------------------------------
__global__ void __launch_bounds__(256)
dssm_build_kernel(const float* __restrict__ log_dt,
                  const float* __restrict__ C,          // (H,N,2) real view
                  const float* __restrict__ log_A_real,
                  const float* __restrict__ A_imag,
                  _Float16* __restrict__ Kshift) {
  __shared__ float sCr[Nn], sCi[Nn], sAr[Nn], sAi[Nn];
  const int h = blockIdx.x;
  const int t = threadIdx.x;

  if (t < Nn) {
    const int n  = t;
    const float dt = __expf(log_dt[h]);
    const float Ar = -__expf(log_A_real[h * Nn + n]);
    const float Ai = A_imag[h * Nn + n];
    const float ar = Ar * dt, ai = Ai * dt;      // dtA
    float s, c;
    __sincosf(ai, &s, &c);
    const float ex = __expf(ar);
    const float Er = ex * c - 1.0f;              // exp(dtA) - 1
    const float Ei = ex * s;
    const float inv = 1.0f / (Ar * Ar + Ai * Ai);
    const float Fr = (Er * Ar + Ei * Ai) * inv;  // (exp(dtA)-1)/A
    const float Fi = (Ei * Ar - Er * Ai) * inv;
    const float Cr = C[(h * Nn + n) * 2 + 0];
    const float Ci = C[(h * Nn + n) * 2 + 1];
    sCr[n] = 2.0f * (Cr * Fr - Ci * Fi);         // fold the 2*Re(...) factor
    sCi[n] = 2.0f * (Cr * Fi + Ci * Fr);
    sAr[n] = ar;
    sAi[n] = ai;
  }

  // Zero this head's Kshift rows (dword stores), then scatter.
  _Float16* dst = Kshift + (size_t)h * KROW;
  unsigned int* dz = (unsigned int*)dst;
  for (int i = t; i < KROW / 2; i += 256) dz[i] = 0u;
  __syncthreads();

  for (int l = t; l < Ll; l += 256) {
    const float fl = (float)l;
    float acc = 0.0f;
    for (int n = 0; n < Nn; ++n) {
      const float e = __expf(sAr[n] * fl);
      float s, c;
      __sincosf(sAi[n] * fl, &s, &c);
      acc = fmaf(e, sCr[n] * c - sCi[n] * s, acc);
    }
    const _Float16 hv = (_Float16)acc;
#pragma unroll
    for (int m = 0; m < 16; ++m) {
      dst[m * KS + (4096 + m - l)] = hv;         // i in [1, 4111] always
    }
  }
}

// ---------------------------------------------------------------------------
// Phase 2: length-masked f16 cast of u, (B,H,L) layout, 4 elems/thread.
// ---------------------------------------------------------------------------
__global__ void __launch_bounds__(256)
dssm_mask_cast(const float* __restrict__ u,
               const int* __restrict__ length,
               _Float16* __restrict__ u16) {
  const int i4 = (blockIdx.x * 256 + threadIdx.x) * 4;  // B*H*L = 2^24
  const int b  = i4 >> 20;                              // H*L = 2^20
  const int l  = i4 & (Ll - 1);
  const int len = length[b];
  const float4 v = *(const float4*)(u + i4);
  v4h o;
  o[0] = (l + 0 < len) ? (_Float16)v.x : (_Float16)0.0f;
  o[1] = (l + 1 < len) ? (_Float16)v.y : (_Float16)0.0f;
  o[2] = (l + 2 < len) ? (_Float16)v.z : (_Float16)0.0f;
  o[3] = (l + 3 < len) ? (_Float16)v.w : (_Float16)0.0f;
  *(v4h*)(u16 + i4) = o;
}

// ---------------------------------------------------------------------------
// Phase 3: y[b,h,l] = sum_{j<=l} K[h,l-j]*u_m[b,h,j] + D[h]*u_m  (masked).
// One wave per 16(l) x 16(batch) tile of head h, lag accumulated 32 at a
// time with v_wmma_f32_16x16x32_f16. A fragment: lane (m=lane&15, hi) loads
// Kshift[h][m][4096 - l0 + 16*hi + 32*js + {0,8} + e] -- 16B-aligned b128s.
// B fragment: u16 column nl, rows hi*16 + [0,16) -- 32B-aligned. Causal
// masking is via Kshift's zero padding; loop count is scalar (readfirstlane)
// so EXEC stays all-ones throughout.
// ---------------------------------------------------------------------------
__global__ void __launch_bounds__(128)
dssm_conv_wmma(const _Float16* __restrict__ Kshift,
               const _Float16* __restrict__ u16,
               const float* __restrict__ u,
               const int* __restrict__ length,
               const float* __restrict__ Dv,
               float* __restrict__ out) {
  const int h    = blockIdx.y;
  const int t    = threadIdx.x;
  const int wave = t >> 5;
  const int lane = t & 31;
  const int hi   = lane >> 4;     // half-wave select
  const int nl   = lane & 15;     // A row (M) / B,C,D column (N = batch)
  const int l0   = (blockIdx.x * 4 + wave) * 16;

  const _Float16* arow = Kshift + (size_t)(h * 16 + nl) * KS
                                + (4096 - l0 + hi * 16);
  const _Float16* urow = u16 + (size_t)(nl * Hh + h) * Ll + hi * 16;

  // Wave-uniform lag-tile count, forced into an SGPR -> scalar loop control,
  // no EXEC manipulation around the WMMAs.
  const int nsteps = __builtin_amdgcn_readfirstlane((l0 + 15) / 32 + 1);

  v8f acc = {};
#pragma unroll 2
  for (int js = 0; js < nsteps; ++js) {
    const int j0 = js * 32;
    union { v16h v; v8h g[2]; } au;
    au.g[0] = *(const v8h*)(arow + j0);           // k = hi*16 + [0,8)
    au.g[1] = *(const v8h*)(arow + j0 + 8);       // k = hi*16 + 8 + [0,8)
    const v16h bf = *(const v16h*)(urow + j0);    // k = hi*16 + [0,16)
    acc = __builtin_amdgcn_wmma_f32_16x16x32_f16(
        /*neg_a=*/false, au.v, /*neg_b=*/false, bf,
        /*c_mod=*/(short)0, acc, /*reuse_a=*/false, /*reuse_b=*/false);
  }

  // Branchless epilogue: + D[h]*u_m, re-apply length mask. Lane holds column
  // nl (batch), rows m = hi*8 + r  ->  l = l0 + hi*8 + r. All loads in-bounds.
  const int len = length[nl];
  const float dh = Dv[h];
  const size_t rowoff = (size_t)(nl * Hh + h) * Ll;
#pragma unroll
  for (int r = 0; r < 8; ++r) {
    const int l = l0 + hi * 8 + r;
    const float uin = u[rowoff + l];
    out[rowoff + l] = (l < len) ? fmaf(dh, uin, acc[r]) : 0.0f;
  }
}

// ---------------------------------------------------------------------------
extern "C" void kernel_launch(void* const* d_in, const int* in_sizes, int n_in,
                              void* d_out, int out_size, void* d_ws, size_t ws_size,
                              hipStream_t stream) {
  const float* u          = (const float*)d_in[0];  // (B,H,L)
  const int*   length     = (const int*)  d_in[1];  // (B,)
  const float* log_dt     = (const float*)d_in[2];  // (H,)
  const float* C          = (const float*)d_in[3];  // (H,N,2)
  const float* log_A_real = (const float*)d_in[4];  // (H,N)
  const float* A_imag     = (const float*)d_in[5];  // (H,N)
  const float* Dv         = (const float*)d_in[6];  // (H,)
  float* out = (float*)d_out;

  _Float16* u16    = (_Float16*)d_ws;                                     // 33.6 MB
  _Float16* Kshift = (_Float16*)((char*)d_ws + (size_t)Bb * Hh * Ll * 2); // 33.8 MB

  dssm_build_kernel<<<Hh, 256, 0, stream>>>(log_dt, C, log_A_real, A_imag,
                                            Kshift);
  dssm_mask_cast<<<(Bb * Hh * Ll) / 1024, 256, 0, stream>>>(u, length, u16);

  dim3 grid(Ll / 64, Hh);   // 4 waves/block, one 16x16 output tile per wave
  dssm_conv_wmma<<<grid, 128, 0, stream>>>(Kshift, u16, u, length, Dv, out);
}
